// EMB_67860483277643
// MI455X (gfx1250) — matrix operand order repaired and checked by reading, not compiled, not run on hardware
//
#include <hip/hip_runtime.h>
#include <hip/hip_bf16.h>
#include <math.h>

// ---------------------------------------------------------------------------
// GNN-attention forward for MI455X (gfx1250, wave32, WMMA).
// Softmax over k of (sq[q]+sk[k]) is q-independent -> attention collapses to a
// single (128,256) node state after round 1; rounds 2..5 are small GEMM pairs.
// Heavy GEMM path: v_wmma_f32_16x16x32_bf16, WG tile 128x128, wave tile 32x64.
// ---------------------------------------------------------------------------

typedef __bf16 bf16_t;
typedef bf16_t v16bf __attribute__((ext_vector_type(16)));
typedef float  v8f   __attribute__((ext_vector_type(8)));

union FragBF { v16bf v; unsigned u[8]; };

__device__ __forceinline__ unsigned pack2_bf16_rne(float a, float b) {
    unsigned ua = __float_as_uint(a);
    unsigned ub = __float_as_uint(b);
    ua += 0x7FFFu + ((ua >> 16) & 1u);
    ub += 0x7FFFu + ((ub >> 16) & 1u);
    return (ua >> 16) | (ub & 0xFFFF0000u);
}

// ---------------------------------------------------------------------------
// Tiled GEMM: C[M,N] = act(A[M,K] @ B[K,N] + bias[N]), f32 in/out, bf16 WMMA.
// Block: 256 threads (8 wave32 waves). WG tile 128x128, wave tile 32x64
// (2x4 frags of 16x16), K-step 32. Requires M%128==0, N%128==0, K%32==0.
// ---------------------------------------------------------------------------
__global__ __launch_bounds__(256)
void wmma_gemm_kernel(const float* __restrict__ A, const float* __restrict__ Bm,
                      const float* __restrict__ bias, float* __restrict__ C,
                      int M, int N, int K, int act) {
    // LDS: A tile 128x32 bf16, row stride 40 halves (20 dwords, conflict-free)
    //      B tile transposed Bt[n][k]: 128x32 bf16, stride 40 halves.
    __shared__ unsigned ldsA[128 * 20];
    __shared__ unsigned ldsB[128 * 20];

    const int tid  = threadIdx.x;
    const int lane = tid & 31;
    const int wave = tid >> 5;
    const int wm   = (wave & 3) * 32;   // 4 waves tile M (128)
    const int wn   = (wave >> 2) * 64;  // 2 waves tile N (128)
    const int row0 = blockIdx.y * 128;
    const int n0   = blockIdx.x * 128;

    v8f acc[2][4];
#pragma unroll
    for (int i = 0; i < 2; ++i)
#pragma unroll
        for (int j = 0; j < 4; ++j)
#pragma unroll
            for (int vv = 0; vv < 8; ++vv) acc[i][j][vv] = 0.0f;

    // Global staging map: per K-step each thread loads 16 A + 16 B floats.
    const int ar = tid >> 1;            // A tile row (0..127)
    const int ah = (tid & 1) * 16;      // 16-float half of the 32-wide K slice
    const int bk = tid >> 3;            // B tile k row (0..31)
    const int bs = (tid & 7) * 16;      // 16-float N segment

    const int lm  = lane & 15;
    const int kha = (lane >= 16) ? 8 : 0;  // A frag K-half offset (dwords)
    const int khb = (lane >= 16) ? 8 : 0;  // B frag K-half offset (dwords)

    for (int k0 = 0; k0 < K; k0 += 32) {
        const float* ap = A + (size_t)(row0 + ar) * K + k0 + ah;
        float4 a0 = *(const float4*)(ap + 0);
        float4 a1 = *(const float4*)(ap + 4);
        float4 a2 = *(const float4*)(ap + 8);
        float4 a3 = *(const float4*)(ap + 12);
        const float* bp = Bm + (size_t)(k0 + bk) * N + n0 + bs;
        float4 b0 = *(const float4*)(bp + 0);
        float4 b1 = *(const float4*)(bp + 4);
        float4 b2 = *(const float4*)(bp + 8);
        float4 b3 = *(const float4*)(bp + 12);
        if (k0 + 32 < K) {                       // WGP-scope prefetch of next K tile
            __builtin_prefetch(ap + 32, 0, 3);   // -> global_prefetch_b8 (WGP)
            __builtin_prefetch(bp + (size_t)32 * N, 0, 3);
        }

        __syncthreads();  // previous compute done before overwriting LDS
        unsigned* aw = &ldsA[ar * 20 + (ah >> 1)];
        aw[0] = pack2_bf16_rne(a0.x, a0.y);
        aw[1] = pack2_bf16_rne(a0.z, a0.w);
        aw[2] = pack2_bf16_rne(a1.x, a1.y);
        aw[3] = pack2_bf16_rne(a1.z, a1.w);
        aw[4] = pack2_bf16_rne(a2.x, a2.y);
        aw[5] = pack2_bf16_rne(a2.z, a2.w);
        aw[6] = pack2_bf16_rne(a3.x, a3.y);
        aw[7] = pack2_bf16_rne(a3.z, a3.w);
        {
            unsigned short* bw = (unsigned short*)ldsB;
            float bf_[16] = {b0.x, b0.y, b0.z, b0.w, b1.x, b1.y, b1.z, b1.w,
                             b2.x, b2.y, b2.z, b2.w, b3.x, b3.y, b3.z, b3.w};
#pragma unroll
            for (int j = 0; j < 16; ++j) {
                unsigned ub = __float_as_uint(bf_[j]);
                ub += 0x7FFFu + ((ub >> 16) & 1u);
                bw[(bs + j) * 40 + bk] = (unsigned short)(ub >> 16);  // Bt[n][k]
            }
        }
        __syncthreads();

        // Fragment loads (4-dword contiguous halves -> ds_load_b128)
        FragBF afr[2], bfr[4];
#pragma unroll
        for (int f = 0; f < 2; ++f) {
            const unsigned* p = &ldsA[(wm + f * 16 + lm) * 20];
#pragma unroll
            for (int vv = 0; vv < 4; ++vv) afr[f].u[vv]     = p[kha + vv];
#pragma unroll
            for (int vv = 0; vv < 4; ++vv) afr[f].u[4 + vv] = p[4 + kha + vv];
        }
#pragma unroll
        for (int f = 0; f < 4; ++f) {
            const unsigned* p = &ldsB[(wn + f * 16 + lm) * 20 + khb];
#pragma unroll
            for (int vv = 0; vv < 8; ++vv) bfr[f].u[vv] = p[vv];
        }

#pragma unroll
        for (int i = 0; i < 2; ++i)
#pragma unroll
            for (int j = 0; j < 4; ++j)
                acc[i][j] = __builtin_amdgcn_wmma_f32_16x16x32_bf16(
                    false, afr[i].v, false, bfr[j].v,
                    (short)0, acc[i][j], false, false);
    }

    // Epilogue: C/D layout -> lane l holds N = l&15; VGPR v holds M = v (+8 hi)
    const int mo = (lane >= 16) ? 8 : 0;
#pragma unroll
    for (int i = 0; i < 2; ++i) {
#pragma unroll
        for (int j = 0; j < 4; ++j) {
            const int n = n0 + wn + j * 16 + lm;
            const float bv = bias ? bias[n] : 0.0f;
#pragma unroll
            for (int vv = 0; vv < 8; ++vv) {
                const int m = row0 + wm + i * 16 + mo + vv;
                float r = acc[i][j][vv] + bv;
                if (act) r = fmaxf(r, 0.0f);
                C[(size_t)m * N + n] = r;
            }
        }
    }
}

// ---------------------------------------------------------------------------
// Fold biases / project attention vectors:
//  block 0: bc[j]  = b_emb @ W_e2h + b_e2h            (j = 0..255)
//  block 1: u[i]   = W_K @ wk  (wk = W_QK[256:512,0]); u[256] = b_K . wk
// ---------------------------------------------------------------------------
__global__ void prep_kernel(const float* __restrict__ b_emb,
                            const float* __restrict__ W_e2h,
                            const float* __restrict__ b_e2h,
                            const float* __restrict__ W_K,
                            const float* __restrict__ b_K,
                            const float* __restrict__ wk,
                            float* __restrict__ bc, float* __restrict__ usk) {
    const int j = threadIdx.x;
    if (blockIdx.x == 0) {
        float a = b_e2h[j];
        for (int i = 0; i < 128; ++i) a += b_emb[i] * W_e2h[i * 256 + j];
        bc[j] = a;
    } else {
        float a = 0.0f;
        for (int c = 0; c < 256; ++c) a += W_K[j * 256 + c] * wk[c];
        usk[j] = a;
        if (j == 0) {
            float cs = 0.0f;
            for (int c = 0; c < 256; ++c) cs += b_K[c] * wk[c];
            usk[256] = cs;
        }
    }
}

// sk[row] = nodes0[row,:] . u + csk   (row = node*128 + batch), one wave/row.
__global__ __launch_bounds__(256)
void sk_kernel(const float* __restrict__ nodes, const float* __restrict__ usk,
               float* __restrict__ sk) {
    const int row  = blockIdx.x * 8 + (threadIdx.x >> 5);
    const int lane = threadIdx.x & 31;
    const float* p = nodes + (size_t)row * 256;
    float a = 0.0f;
    for (int j = lane; j < 256; j += 32) a += p[j] * usk[j];
    for (int off = 16; off; off >>= 1) a += __shfl_xor(a, off, 32);
    if (lane == 0) sk[row] = a + usk[256];
}

// ctx[b,h] = sum_n softmax_n(sk[n,b]) * nodes0[n*128+b, h]   (one block per b)
__global__ __launch_bounds__(256)
void attn_ctx_kernel(const float* __restrict__ sk, const float* __restrict__ nodes,
                     float* __restrict__ ctx) {
    const int b = blockIdx.x, tid = threadIdx.x;
    __shared__ float w[512];
    __shared__ float red[256];

    const float s0 = sk[tid * 128 + b];
    const float s1 = sk[(tid + 256) * 128 + b];

    red[tid] = fmaxf(s0, s1);
    __syncthreads();
    for (int s = 128; s; s >>= 1) {
        if (tid < s) red[tid] = fmaxf(red[tid], red[tid + s]);
        __syncthreads();
    }
    const float mx = red[0];
    __syncthreads();

    const float e0 = __expf(s0 - mx), e1 = __expf(s1 - mx);
    red[tid] = e0 + e1;
    __syncthreads();
    for (int s = 128; s; s >>= 1) {
        if (tid < s) red[tid] += red[tid + s];
        __syncthreads();
    }
    const float inv = 1.0f / red[0];
    w[tid] = e0 * inv;
    w[tid + 256] = e1 * inv;
    __syncthreads();

    float a = 0.0f;  // h = tid; coalesced 1KB reads per n, nodes0 is L2-resident
    for (int n = 0; n < 512; ++n)
        a += w[n] * nodes[((size_t)(n * 128 + b)) * 256 + tid];
    ctx[b * 256 + tid] = a;
}

// cat[b, 0:128] = embB[b]; cat[b, 128:384] = agg[b]
__global__ void concat_kernel(const float* __restrict__ embB,
                              const float* __restrict__ agg,
                              float* __restrict__ cat) {
    const int idx = blockIdx.x * 256 + threadIdx.x;  // 128*384 total
    const int b = idx / 384, j = idx % 384;
    cat[idx] = (j < 128) ? embB[b * 128 + j] : agg[b * 256 + (j - 128)];
}

// logits = h @ W_fc2 + b_fc2 ; out = log_softmax(logits) over 2 classes
__global__ void head_kernel(const float* __restrict__ h, const float* __restrict__ Wf,
                            const float* __restrict__ bf, float* __restrict__ out) {
    const int b = threadIdx.x;  // 128
    float l0 = bf[0], l1 = bf[1];
    for (int i = 0; i < 256; ++i) {
        const float x = h[b * 256 + i];
        l0 += x * Wf[i * 2 + 0];
        l1 += x * Wf[i * 2 + 1];
    }
    const float m = fmaxf(l0, l1);
    const float lse = m + logf(__expf(l0 - m) + __expf(l1 - m));
    out[b * 2 + 0] = l0 - lse;
    out[b * 2 + 1] = l1 - lse;
}

// ---------------------------------------------------------------------------
extern "C" void kernel_launch(void* const* d_in, const int* in_sizes, int n_in,
                              void* d_out, int out_size, void* d_ws, size_t ws_size,
                              hipStream_t stream) {
    (void)in_sizes; (void)n_in; (void)out_size; (void)ws_size;
    const float* As    = (const float*)d_in[0];   // (512,128,128)
    const float* Bin   = (const float*)d_in[1];   // (128,128)
    const float* W_emb = (const float*)d_in[2];
    const float* b_emb = (const float*)d_in[3];
    const float* W_e2h = (const float*)d_in[4];
    const float* b_e2h = (const float*)d_in[5];
    // d_in[6] W_Q, d_in[7] b_Q: unused (sq cancels in softmax over k)
    const float* W_K   = (const float*)d_in[8];
    const float* b_K   = (const float*)d_in[9];
    const float* W_V   = (const float*)d_in[10];
    const float* b_V   = (const float*)d_in[11];
    const float* W_QK  = (const float*)d_in[12];  // wk = W_QK + 256
    // d_in[13] b_QK: cancels
    const float* W_A2N = (const float*)d_in[14];
    const float* b_A2N = (const float*)d_in[15];
    const float* W_one = (const float*)d_in[16];
    const float* b_one = (const float*)d_in[17];
    const float* W_fc1 = (const float*)d_in[18];
    const float* b_fc1 = (const float*)d_in[19];
    const float* W_fc2 = (const float*)d_in[20];
    const float* b_fc2 = (const float*)d_in[21];
    float* out = (float*)d_out;
    char*  ws  = (char*)d_ws;

    // Workspace layout (256B aligned)
    auto al = [](size_t x) { return (x + 255) & ~(size_t)255; };
    size_t o = 0;
    float* Wc    = (float*)(ws + o); o = al(o + (size_t)128 * 256 * 4);   // W_emb@W_e2h
    float* bc    = (float*)(ws + o); o = al(o + 256 * 4);
    float* usk   = (float*)(ws + o); o = al(o + 512 * 4);                 // u[256], csk
    float* nodes = (float*)(ws + o); o = al(o + (size_t)65536 * 256 * 4); // 64 MB
    float* sk    = (float*)(ws + o); o = al(o + (size_t)65536 * 4);
    float* ctx   = (float*)(ws + o); o = al(o + (size_t)128 * 256 * 4);
    float* tmpv  = (float*)(ws + o); o = al(o + (size_t)128 * 256 * 4);
    float* nA    = (float*)(ws + o); o = al(o + (size_t)128 * 256 * 4);
    float* nB    = (float*)(ws + o); o = al(o + (size_t)128 * 256 * 4);
    float* embB  = (float*)(ws + o); o = al(o + (size_t)128 * 128 * 4);
    float* cat   = (float*)(ws + o); o = al(o + (size_t)128 * 384 * 4);
    float* hbuf  = (float*)(ws + o); o = al(o + (size_t)128 * 256 * 4);

    // 1) Fold constants: bc, u/csk; Wc = W_emb @ W_e2h (no intervening nonlinearity)
    prep_kernel<<<2, 256, 0, stream>>>(b_emb, W_e2h, b_e2h, W_K, b_K, W_QK + 256, bc, usk);
    wmma_gemm_kernel<<<dim3(2, 1), 256, 0, stream>>>(W_emb, W_e2h, nullptr, Wc, 128, 256, 128, 0);

    // 2) nodes0 = relu(As @ Wc + bc)  -- the one big GEMM (65536x256x128)
    wmma_gemm_kernel<<<dim3(2, 512), 256, 0, stream>>>(As, Wc, bc, nodes, 65536, 256, 128, 1);

    // 3) Attention round 1 (q cancels): sk = nodes0@u + csk; ctx = softmax-weighted sum
    sk_kernel<<<8192, 256, 0, stream>>>(nodes, usk, sk);
    attn_ctx_kernel<<<128, 256, 0, stream>>>(sk, nodes, ctx);

    // 4) Rounds 1..5 collapse to: N = relu((N@W_V + b_V)@W_A2N + b_A2N), N := ctx
    const float* cur = ctx;
    float* bufs[2] = {nA, nB};
    for (int r = 0; r < 5; ++r) {
        wmma_gemm_kernel<<<dim3(2, 1), 256, 0, stream>>>(cur, W_V, b_V, tmpv, 128, 256, 256, 0);
        wmma_gemm_kernel<<<dim3(2, 1), 256, 0, stream>>>(tmpv, W_A2N, b_A2N, bufs[r & 1], 128, 256, 256, 1);
        cur = bufs[r & 1];
    }
    // max over 512 identical nodes == cur

    // 5) Head: embB = B@W_one + b_one; h = relu(concat@W_fc1 + b_fc1); log_softmax
    wmma_gemm_kernel<<<dim3(1, 1), 256, 0, stream>>>(Bin, W_one, b_one, embB, 128, 128, 128, 0);
    concat_kernel<<<192, 256, 0, stream>>>(embB, cur, cat);
    wmma_gemm_kernel<<<dim3(2, 1), 256, 0, stream>>>(cat, W_fc1, b_fc1, hbuf, 128, 256, 384, 1);
    head_kernel<<<1, 128, 0, stream>>>(hbuf, W_fc2, b_fc2, out);
}
